// OutputLayer_65584150610613
// MI455X (gfx1250) — matrix operand order repaired and checked by reading, not compiled, not run on hardware
//
#include <hip/hip_runtime.h>
#include <hip/hip_bf16.h>
#include <stdint.h>

// Problem constants (from reference)
#define B_   8
#define T_   4000
#define D_   512
#define L_   8922
#define LP_  8960          // L padded to multiple of LT_
#define LT_  128           // labels per block (8 waves x 16)
#define TT_  80            // timesteps per t-tile (5 WMMA n-tiles of 16); 4000 % 80 == 0
#define KT_  32            // bf16 WMMA K per step; 512 / 32 = 16 chunks
#define NK_  (D_ / KT_)    // 16 k-chunks
#define NT_  (T_ / TT_)    // 50 t-tiles
#define NIT_ (NK_ * NT_)   // 800 pipelined iterations
#define ROWS 40            // padded LDS row stride in bf16 elems (80B = 64B data + 16B TDM pad)
                           // bank of lane m's b128 = (m*20)%64 -> all 16 lanes distinct banks

typedef __attribute__((ext_vector_type(16))) __bf16        v16bf;
typedef __attribute__((ext_vector_type(8)))  float         v8f;
typedef __attribute__((ext_vector_type(4)))  unsigned int  v4u;
typedef __attribute__((ext_vector_type(8)))  int           v8i_t;
typedef __attribute__((ext_vector_type(4)))  int           v4i_t;

struct U8x32 { uint4 a, b; };   // 32 bytes == v16bf

#if defined(__has_builtin)
#if __has_builtin(__builtin_amdgcn_tensor_load_to_lds) && __has_builtin(__builtin_amdgcn_s_wait_tensorcnt)
#define USE_TDM 1
#endif
#endif
#ifndef USE_TDM
#define USE_TDM 0
#endif

__device__ __forceinline__ unsigned short f2bf(float f) {
  unsigned u = __builtin_bit_cast(unsigned, f);
  u += 0x7FFFu + ((u >> 16) & 1u);          // round-to-nearest-even
  return (unsigned short)(u >> 16);
}

// ---------------- prep: f32 -> bf16 conversions ----------------
__global__ void cvt_x_kernel(const float* __restrict__ x,
                             unsigned short* __restrict__ xb, int n4) {
  int i = blockIdx.x * blockDim.x + threadIdx.x;
  if (i >= n4) return;
  const float4 f = ((const float4*)x)[i];
  uint2 o;
  o.x = (unsigned)f2bf(f.x) | ((unsigned)f2bf(f.y) << 16);
  o.y = (unsigned)f2bf(f.z) | ((unsigned)f2bf(f.w) << 16);
  ((uint2*)xb)[i] = o;
}

__global__ void prep_w_kernel(const float* __restrict__ U,
                              const float* __restrict__ F,
                              const float* __restrict__ P,
                              unsigned short* __restrict__ Ub,
                              unsigned short* __restrict__ Wb, int n4) {
  int i = blockIdx.x * blockDim.x + threadIdx.x;
  if (i >= n4) return;
  int e = i * 4;
  int l = e / D_;
  int d = e % D_;
  uint2 ou = {0u, 0u}, ow = {0u, 0u};
  if (l < L_) {
    const float4 u = *(const float4*)(U + (size_t)l * D_ + d);
    const float4 f = *(const float4*)(F + (size_t)l * D_ + d);
    const float4 p = *(const float4*)(P + d);
    ou.x = (unsigned)f2bf(u.x)       | ((unsigned)f2bf(u.y)       << 16);
    ou.y = (unsigned)f2bf(u.z)       | ((unsigned)f2bf(u.w)       << 16);
    ow.x = (unsigned)f2bf(f.x * p.x) | ((unsigned)f2bf(f.y * p.y) << 16);
    ow.y = (unsigned)f2bf(f.z * p.z) | ((unsigned)f2bf(f.w * p.w) << 16);
  }
  ((uint2*)Ub)[i] = ou;
  ((uint2*)Wb)[i] = ow;
}

// ---------------- TDM: 2D tile load, Global -> LDS, bank-depad ----------------
// D# per CDNA5 ISA ch.8.3-8.5. Row-major tile [tile_d1][tile_d0] lands in LDS
// with 16B padding after every 64B row (pad_interval=16 DW, pad_amount=4 DW),
// giving the conflict-free 80B row stride the fragment reads use.
#if USE_TDM
__device__ __forceinline__ void tdm_load_2d(unsigned lds_byte_addr,
                                            const unsigned short* gptr,
                                            unsigned tile_d0, unsigned tile_d1,
                                            unsigned tensor_d0, unsigned tensor_d1,
                                            unsigned stride_d0) {
  unsigned long long ga = (unsigned long long)(uintptr_t)gptr;
  v4u g0;
  g0[0] = 1u;                                      // count=1 (valid), user mode
  g0[1] = lds_byte_addr;                           // [63:32]  lds_addr
  g0[2] = (unsigned)(ga & 0xFFFFFFFFu);            // [95:64]  global_addr lo
  g0[3] = (unsigned)((ga >> 32) & 0x01FFFFFFu)     // [120:96] global_addr hi
        | (2u << 30);                              // [127:126] type=2 (image)
  v8i_t g1;
  g1[0] = (int)((1u << 16)                         // data_size=1 -> 2 bytes
        |        (1u << 20)                        // pad_enable
        |        (3u << 22)                        // pad_interval: 16 DWORDs (64B)
        |        (3u << 25));                      // pad_amount:   4 DWORDs (16B)
  g1[1] = (int)((tensor_d0 & 0xFFFFu) << 16);      // [63:48]  tensor_dim0 lo16
  g1[2] = (int)(((tensor_d0 >> 16) & 0xFFFFu)      // [79:64]  tensor_dim0 hi16
        |        ((tensor_d1 & 0xFFFFu) << 16));   // [95:80]  tensor_dim1 lo16
  g1[3] = (int)(((tensor_d1 >> 16) & 0xFFFFu)      // [111:96] tensor_dim1 hi16
        |        ((tile_d0 & 0xFFFFu) << 16));     // [127:112] tile_dim0
  g1[4] = (int)(tile_d1 & 0xFFFFu);                // [143:128] tile_dim1; tile_dim2=0
  g1[5] = (int)stride_d0;                          // [191:160] tensor_dim0_stride lo32
  g1[6] = 0;                                       // stride hi16 / dim1_stride lo16
  g1[7] = 0;
  v4i_t z4 = {0, 0, 0, 0};
#if __clang_major__ >= 23
  v8i_t z8 = {0, 0, 0, 0, 0, 0, 0, 0};
  __builtin_amdgcn_tensor_load_to_lds(g0, g1, z4, z4, z8, 0);
#else
  __builtin_amdgcn_tensor_load_to_lds(g0, g1, z4, z4, 0);
#endif
}
#endif

// ---------------- fused dual-GEMM + product-reduce ----------------
// logits[b,l] = sum_t (U_w[l,:].x[b,t,:]) * (w2[l,:].x[b,t,:]) + final_b[l]
struct SmemBuf {
  unsigned short uT[LT_ * ROWS];   // 10240 B (padded rows)
  unsigned short wT[LT_ * ROWS];   // 10240 B
  unsigned short xT[TT_ * ROWS];   //  6400 B
};
#define OFF_U 0u
#define OFF_W (unsigned)(LT_ * ROWS * 2)
#define OFF_X (unsigned)(LT_ * ROWS * 4)

__global__ __launch_bounds__(256) void fused_attn_logits(
    const unsigned short* __restrict__ xb,   // [B][T][D] bf16
    const unsigned short* __restrict__ Ub,   // [LP][D]  bf16
    const unsigned short* __restrict__ Wb,   // [LP][D]  bf16 (final_w * proj_w)
    const float* __restrict__ final_b,       // [L]
    float* __restrict__ out)                 // [B][L]
{
  __shared__ SmemBuf sm[2];                  // 2 x 26.9 KB, double buffered

  const int tid  = threadIdx.x;
  const int lane = tid & 31;
  const int wave = tid >> 5;               // 0..7
  const int lblk = blockIdx.x * LT_;       // label tile base
  const int b    = blockIdx.y;

  const int mrow = wave * 16;              // this wave's 16-label row in the tile
  const int am   = mrow + (lane & 15);     // A-frag row (16-bit A 16x32 layout)
  const int akb  = (lane < 16) ? 0 : 8;    // A: lanes 0-15 K 0..7/16..23; 16-31 K 8..15/24..31
  const int xkb  = (lane < 16) ? 0 : 8;    // B: lanes 0-15 K 0..7/16..23; 16-31 K 8..15/24..31

  const unsigned short* xbb = xb + (size_t)b * T_ * D_;
  // Low 32 bits of the flat shared-memory address == LDS byte offset
  // (flat aperture mapping).
  const unsigned ldsbase = (unsigned)(uintptr_t)(void*)&sm[0];

  float plog[8];
#pragma unroll
  for (int r = 0; r < 8; ++r) plog[r] = 0.f;
  v8f accA[5], accC[5];

#if USE_TDM
  // One wave drives the Tensor Data Mover; descriptors are pure SGPR data.
  auto issue = [&](int itn) {
    if (wave != 0) return;
    const int tt = itn / NK_;
    const int kk = itn & (NK_ - 1);
    const unsigned base = ldsbase + (unsigned)((itn & 1) * (int)sizeof(SmemBuf));
    const size_t koff = (size_t)(kk * KT_);
    tdm_load_2d(base + OFF_U, Ub  + (size_t)lblk * D_ + koff, KT_, LT_, D_, LP_, D_);
    tdm_load_2d(base + OFF_W, Wb  + (size_t)lblk * D_ + koff, KT_, LT_, D_, LP_, D_);
    tdm_load_2d(base + OFF_X, xbb + (size_t)(tt * TT_) * D_ + koff, KT_, TT_, D_, T_, D_);
  };
  issue(0);                                // prologue: fill buffer 0
#endif

  for (int it = 0; it < NIT_; ++it) {
    const int kk  = it & (NK_ - 1);
    const SmemBuf& S = sm[it & 1];

#if USE_TDM
    if (it + 1 < NIT_) {
      issue(it + 1);                                   // prefetch next chunk
      __builtin_amdgcn_s_wait_tensorcnt(3);            // current chunk's 3 loads done
    } else {
      __builtin_amdgcn_s_wait_tensorcnt(0);            // drain
    }
    __syncthreads();                                   // data visible to all waves
#else
    // Fallback: synchronous cooperative staging into the same padded layout.
    {
      const int tt = it / NK_;
      unsigned short* du = sm[it & 1].uT;
      unsigned short* dw = sm[it & 1].wT;
      for (int c = tid; c < (LT_ * KT_) / 8; c += 256) {
        int row = c >> 2;
        int col = (c & 3) << 3;
        size_t g = (size_t)(lblk + row) * D_ + (size_t)(kk * KT_ + col);
        *(uint4*)(du + row * ROWS + col) = *(const uint4*)(Ub + g);
        *(uint4*)(dw + row * ROWS + col) = *(const uint4*)(Wb + g);
      }
      unsigned short* dx = sm[it & 1].xT;
      for (int c = tid; c < (TT_ * KT_) / 8; c += 256) {
        int row = c >> 2;
        int col = (c & 3) << 3;
        size_t g = (size_t)(tt * TT_ + row) * D_ + (size_t)(kk * KT_ + col);
        *(uint4*)(dx + row * ROWS + col) = *(const uint4*)(xbb + g);
      }
    }
    __syncthreads();
#endif

    if (kk == 0) {
#pragma unroll
      for (int n = 0; n < 5; ++n) {
        accA[n] = (v8f){0.f,0.f,0.f,0.f,0.f,0.f,0.f,0.f};
        accC[n] = (v8f){0.f,0.f,0.f,0.f,0.f,0.f,0.f,0.f};
      }
    }

    // -------- preload ALL fragments, then drain WMMAs (fewer dscnt stalls) --
    v16bf bX[5];
#pragma unroll
    for (int n = 0; n < 5; ++n) {
      const uint4* px = (const uint4*)&S.xT[(n * 16 + (lane & 15)) * ROWS + xkb];
      bX[n] = __builtin_bit_cast(v16bf, U8x32{px[0], px[2]});  // K 0..7 | 16..23 per half
    }
    const uint4* pu = (const uint4*)&S.uT[am * ROWS + akb];
    const uint4* pw = (const uint4*)&S.wT[am * ROWS + akb];
    v16bf aU = __builtin_bit_cast(v16bf, U8x32{pu[0], pu[2]});  // K 0..7 | 16..23 per half
    v16bf aW = __builtin_bit_cast(v16bf, U8x32{pw[0], pw[2]});

#pragma unroll
    for (int n = 0; n < 5; ++n) {
      accA[n] = __builtin_amdgcn_wmma_f32_16x16x32_bf16(
          false, aU, false, bX[n], (short)0, accA[n], false, false);
      accC[n] = __builtin_amdgcn_wmma_f32_16x16x32_bf16(
          false, aW, false, bX[n], (short)0, accC[n], false, false);
    }

    if (kk == NK_ - 1) {
      // K-sums complete for this t-tile: elementwise product, accumulate over N
#pragma unroll
      for (int n = 0; n < 5; ++n)
#pragma unroll
        for (int r = 0; r < 8; ++r)
          plog[r] += accA[n][r] * accC[n][r];
    }
    __syncthreads();   // buffer free before TDM refills it next iteration
  }

  // C/D layout: VGPR r -> M=r (lanes 0-15) / M=r+8 (lanes 16-31); N = lane%16.
  // Reduce over N = 16-lane groups, then one lane per group writes 8 labels.
  const int half = lane >> 4;
#pragma unroll
  for (int r = 0; r < 8; ++r) {
    float s = plog[r];
    s += __shfl_xor(s, 1, 16);
    s += __shfl_xor(s, 2, 16);
    s += __shfl_xor(s, 4, 16);
    s += __shfl_xor(s, 8, 16);
    if ((lane & 15) == 0) {
      int label = lblk + mrow + half * 8 + r;
      if (label < L_) out[(size_t)b * L_ + label] = s + final_b[label];
    }
  }
}

extern "C" void kernel_launch(void* const* d_in, const int* in_sizes, int n_in,
                              void* d_out, int out_size, void* d_ws, size_t ws_size,
                              hipStream_t stream) {
  const float* x  = (const float*)d_in[0];   // [B,T,D]
  const float* Uw = (const float*)d_in[1];   // [L,D]
  const float* Fw = (const float*)d_in[2];   // [L,D]
  const float* fb = (const float*)d_in[3];   // [L]
  const float* pw = (const float*)d_in[4];   // [D]
  float* out = (float*)d_out;                // [B,L]

  unsigned short* xb = (unsigned short*)d_ws;            // B*T*D bf16 (32.8 MB)
  unsigned short* Ub = xb + (size_t)B_ * T_ * D_;        // LP*D bf16 (9.2 MB)
  unsigned short* Wb = Ub + (size_t)LP_ * D_;            // LP*D bf16 (9.2 MB)

  const int n4x = (B_ * T_ * D_) / 4;
  cvt_x_kernel<<<(n4x + 255) / 256, 256, 0, stream>>>(x, xb, n4x);

  const int n4w = (LP_ * D_) / 4;
  prep_w_kernel<<<(n4w + 255) / 256, 256, 0, stream>>>(Uw, Fw, pw, Ub, Wb, n4w);

  dim3 grid(LP_ / LT_, B_);   // 70 label tiles x 8 batches = 560 blocks
  fused_attn_logits<<<grid, 256, 0, stream>>>(xb, Ub, Wb, fb, out);
}
